// PerResidueLDDTCaPredictor_86148454023176
// MI455X (gfx1250) — matrix operand order repaired and checked by reading, not compiled, not run on hardware
//
#include <hip/hip_runtime.h>
#include <hip/hip_bf16.h>

// ---------------- problem constants ----------------
#define Bc     2
#define Lc     256
#define CINc   1024
#define Cc     128
#define Hc     8
#define NBc    50
#define NZc    64
#define NLAYERc 8
#define DBINc  128
#define DHc    16
#define BLc    (Bc*Lc)          // 512
#define BLLc   (Bc*Lc*Lc)       // 131072
#define PAIRc  ((size_t)BLLc*Cc)// 16,777,216

// ---------------- types ----------------
typedef __attribute__((ext_vector_type(16))) __bf16 v16bf;
typedef __attribute__((ext_vector_type(8)))  __bf16 v8bf;
typedef __attribute__((ext_vector_type(8)))  float  v8f;
typedef __attribute__((ext_vector_type(4)))  unsigned int v4u;
typedef __attribute__((ext_vector_type(8)))  int v8i32;
typedef __attribute__((ext_vector_type(4)))  int v4i32;

static __device__ __forceinline__ __bf16 f2bf(float f) {
    unsigned u = __builtin_bit_cast(unsigned, f);
    unsigned r = (u + 0x7FFFu + ((u >> 16) & 1u)) >> 16;   // round-to-nearest-even
    return __builtin_bit_cast(__bf16, (unsigned short)r);
}
static __device__ __forceinline__ float bf2f(__bf16 b) {
    unsigned u = ((unsigned)__builtin_bit_cast(unsigned short, b)) << 16;
    return __builtin_bit_cast(float, u);
}
static __device__ __forceinline__ float silu(float x) {
    return x / (1.0f + __expf(-x));
}

// load a 16-half WMMA fragment (two contiguous 8-half chunks)
static __device__ __forceinline__ v16bf load_frag(const __bf16* p0, const __bf16* p1) {
    v8bf lo = *(const v8bf*)p0;
    v8bf hi = *(const v8bf*)p1;
    v16bf f;
#pragma unroll
    for (int i = 0; i < 8; ++i) { f[i] = lo[i]; f[i + 8] = hi[i]; }
    return f;
}

// ---------------- generic bf16 WMMA GEMM (direct-global weights) ----------------
// out[M,N] = A[M,K] (bf16, row-major) @ Wt[N,K] (bf16, pre-transposed) + bias
// EPI: 0 = f32 out; 1 = f32 out + residual add; 2 = bf16 out with SiLU
template <int NT, int EPI>
__global__ __launch_bounds__(256)
void k_gemm(const __bf16* __restrict__ A, const __bf16* __restrict__ Wt,
            const float* __restrict__ bias, const float* __restrict__ res,
            float* __restrict__ outf, __bf16* __restrict__ outb, int K) {
    const int wave = threadIdx.x >> 5;
    const int lane = threadIdx.x & 31;
    const int r    = lane & 15;
    const int kg   = lane >> 4;
    const int rowbase = blockIdx.x * 128 + wave * 16;

    v8f acc[NT];
#pragma unroll
    for (int nt = 0; nt < NT; ++nt)
#pragma unroll
        for (int j = 0; j < 8; ++j) acc[nt][j] = 0.0f;

    const __bf16* arow = A + (size_t)(rowbase + r) * K;
    for (int kt = 0; kt < K; kt += 32) {
        v16bf af = load_frag(arow + kt + kg * 16, arow + kt + kg * 16 + 8);
#pragma unroll
        for (int nt = 0; nt < NT; ++nt) {
            const __bf16* bcol = Wt + (size_t)(nt * 16 + r) * K + kt + kg * 16;
            v16bf bfrag = load_frag(bcol, bcol + 8);
            acc[nt] = __builtin_amdgcn_wmma_f32_16x16x32_bf16(
                false, af, false, bfrag, (short)0, acc[nt], false, false);
        }
    }
#pragma unroll
    for (int nt = 0; nt < NT; ++nt) {
        const int ocol = nt * 16 + r;
        const float bv = bias[ocol];
#pragma unroll
        for (int rr = 0; rr < 8; ++rr) {
            const int orow = rowbase + rr + kg * 8;
            const size_t oi = (size_t)orow * (NT * 16) + ocol;
            float vv = acc[nt][rr] + bv;
            if (EPI == 0)      outf[oi] = vv;
            else if (EPI == 1) outf[oi] = res[oi] + vv;
            else               outb[oi] = f2bf(silu(vv));
        }
    }
}

// ---------------- TDM-staged bf16 WMMA GEMM (K=128, weights DMA'd to LDS) ----------------
template <int NT, int EPI>
__global__ __launch_bounds__(256)
void k_gemm_tdm(const __bf16* __restrict__ A, const __bf16* __restrict__ Wt,
                const float* __restrict__ bias, const float* __restrict__ res,
                float* __restrict__ outf, __bf16* __restrict__ outb) {
    constexpr int K = 128;
    __shared__ __bf16 lw[NT * 16 * K];        // single LDS object -> offset 0
    const int wave = threadIdx.x >> 5;
    const int lane = threadIdx.x & 31;
    const int r    = lane & 15;
    const int kg   = lane >> 4;
    const int rowbase = blockIdx.x * 128 + wave * 16;

    if (wave == 0) {
        // Tensor Data Mover: DMA Wt tile (NT*16 rows x 128 cols, bf16) -> LDS@0
        unsigned long long ga = (unsigned long long)(const void*)Wt;
        const unsigned td0 = K, td1 = NT * 16, tile0 = K, tile1 = NT * 16;
        v4u g0;
        g0[0] = 1u;                                             // count = 1 (valid D#)
        g0[1] = 0u;                                             // lds_addr = 0
        g0[2] = (unsigned)(ga & 0xFFFFFFFFu);                   // global_addr lo
        g0[3] = (unsigned)((ga >> 32) & 0x01FFFFFFu) | 0x80000000u; // addr hi | type=2
        v8i32 g1;
        g1[0] = (int)(1u << 16);                                // data_size = 2 bytes
        g1[1] = (int)((td0 & 0xFFFFu) << 16);                   // tensor_dim0[15:0]
        g1[2] = (int)((td0 >> 16) | ((td1 & 0xFFFFu) << 16));   // td0 hi | td1 lo
        g1[3] = (int)((td1 >> 16) | (tile0 << 16));             // td1 hi | tile_dim0
        g1[4] = (int)tile1;                                     // tile_dim1 (tile_dim2=0)
        g1[5] = (int)td0;                                       // tensor_dim0_stride lo
        g1[6] = 0; g1[7] = 0;
        v4i32 gz = {0, 0, 0, 0};
#if __clang_major__ >= 23
        v8i32 g4 = {0, 0, 0, 0, 0, 0, 0, 0};
        __builtin_amdgcn_tensor_load_to_lds(g0, g1, gz, gz, g4, 0);
#else
        __builtin_amdgcn_tensor_load_to_lds(g0, g1, gz, gz, 0);
#endif
        __builtin_amdgcn_s_wait_tensorcnt(0);
    }
    __syncthreads();

    v8f acc[NT];
#pragma unroll
    for (int nt = 0; nt < NT; ++nt)
#pragma unroll
        for (int j = 0; j < 8; ++j) acc[nt][j] = 0.0f;

    const __bf16* arow = A + (size_t)(rowbase + r) * K;
#pragma unroll
    for (int kt = 0; kt < K; kt += 32) {
        v16bf af = load_frag(arow + kt + kg * 16, arow + kt + kg * 16 + 8);
#pragma unroll
        for (int nt = 0; nt < NT; ++nt) {
            const __bf16* bcol = &lw[(nt * 16 + r) * K + kt + kg * 16];
            v16bf bfrag = load_frag(bcol, bcol + 8);
            acc[nt] = __builtin_amdgcn_wmma_f32_16x16x32_bf16(
                false, af, false, bfrag, (short)0, acc[nt], false, false);
        }
    }
#pragma unroll
    for (int nt = 0; nt < NT; ++nt) {
        const int ocol = nt * 16 + r;
        const float bv = bias[ocol];
#pragma unroll
        for (int rr = 0; rr < 8; ++rr) {
            const int orow = rowbase + rr + kg * 8;
            const size_t oi = (size_t)orow * (NT * 16) + ocol;
            float vv = acc[nt][rr] + bv;
            if (EPI == 0)      outf[oi] = vv;
            else if (EPI == 1) outf[oi] = res[oi] + vv;
            else               outb[oi] = f2bf(silu(vv));
        }
    }
}

// ---------------- LayerNorm (sequence, C=128) -> bf16 ----------------
__global__ void k_ln_bf16(const float* __restrict__ x, const float* __restrict__ g,
                          const float* __restrict__ b, __bf16* __restrict__ out) {
    const size_t row = blockIdx.x;
    const int c = threadIdx.x;
    __shared__ float red[Cc];
    float v = x[row * Cc + c];
    red[c] = v; __syncthreads();
    for (int off = 64; off > 0; off >>= 1) { if (c < off) red[c] += red[c + off]; __syncthreads(); }
    float mean = red[0] * (1.0f / Cc); __syncthreads();
    float d = v - mean;
    red[c] = d * d; __syncthreads();
    for (int off = 64; off > 0; off >>= 1) { if (c < off) red[c] += red[c + off]; __syncthreads(); }
    float inv = __frsqrt_rn(red[0] * (1.0f / Cc) + 1e-5f);
    out[row * Cc + c] = f2bf(d * inv * g[c] + b[c]);
}

// ---------------- fused rank-1 pair update + pair LayerNorm (wave per row) ----------------
__global__ __launch_bounds__(256)
void k_pairupd_ln(float* __restrict__ xpair, const float* __restrict__ a,
                  const float* __restrict__ bpp, const float* __restrict__ g,
                  const float* __restrict__ bta, __bf16* __restrict__ out) {
    const size_t wid = (size_t)blockIdx.x * 8 + (threadIdx.x >> 5);  // (b,l,m) flat
    const int lane = threadIdx.x & 31;
    const int m = (int)(wid % Lc);
    const size_t bl = wid / Lc;
    const int l = (int)(bl % Lc), b = (int)(bl / Lc);
    float* row = xpair + wid * Cc;
    const float* arow = a   + (size_t)(b * Lc + l) * Cc;
    const float* brow = bpp + (size_t)(b * Lc + m) * Cc;
    float v[4];
    float s = 0.0f;
#pragma unroll
    for (int j = 0; j < 4; ++j) {
        const int c = lane + 32 * j;
        v[j] = row[c] + arow[c] * brow[c];
        row[c] = v[j];
        s += v[j];
    }
    for (int off = 16; off > 0; off >>= 1) s += __shfl_down(s, off, 32);
    const float mean = __shfl(s, 0, 32) * (1.0f / Cc);
    float var = 0.0f;
#pragma unroll
    for (int j = 0; j < 4; ++j) { float d = v[j] - mean; var += d * d; }
    for (int off = 16; off > 0; off >>= 1) var += __shfl_down(var, off, 32);
    const float inv = __frsqrt_rn(__shfl(var, 0, 32) * (1.0f / Cc) + 1e-5f);
#pragma unroll
    for (int j = 0; j < 4; ++j) {
        const int c = lane + 32 * j;
        out[wid * Cc + c] = f2bf((v[j] - mean) * inv * g[c] + bta[c]);
    }
}

// ---------------- converters ----------------
__global__ void k_f2b(const float* __restrict__ s, __bf16* __restrict__ d, size_t n) {
    for (size_t i = (size_t)blockIdx.x * blockDim.x + threadIdx.x; i < n;
         i += (size_t)gridDim.x * blockDim.x) d[i] = f2bf(s[i]);
}
// batched transpose+convert: src [NL,K,N] f32 -> dst [NL,N,K] bf16
__global__ void k_wt(const float* __restrict__ s, __bf16* __restrict__ d, int K, int N, int NL) {
    const size_t tot = (size_t)NL * K * N;
    for (size_t i = (size_t)blockIdx.x * blockDim.x + threadIdx.x; i < tot;
         i += (size_t)gridDim.x * blockDim.x) {
        size_t nl = i / ((size_t)K * N);
        size_t rm = i % ((size_t)K * N);
        int k = (int)(rm / N), n = (int)(rm % N);
        d[nl * (size_t)K * N + (size_t)n * K + k] = f2bf(s[i]);
    }
}

// ---------------- distance embedding init ----------------
__global__ void k_dist(const float* __restrict__ pos, const float* __restrict__ table,
                       float* __restrict__ xpair, float* __restrict__ biasout) {
    const int id = blockIdx.x;                 // (b,l,m)
    const int m = id % Lc, l = (id / Lc) % Lc, b = id / (Lc * Lc);
    const float* pl = pos + (b * Lc + l) * 3;
    const float* pm = pos + (b * Lc + m) * 3;
    float dx = pl[0] - pm[0], dy = pl[1] - pm[1], dz = pl[2] - pm[2];
    float dist = sqrtf(dx * dx + dy * dy + dz * dz + 1e-12f);
    if (threadIdx.x == 0) biasout[id] = dist;
    int idx = (int)floorf(dist * 4.0f);
    idx = idx < 0 ? 0 : (idx > DBINc - 1 ? DBINc - 1 : idx);
    xpair[(size_t)id * Cc + threadIdx.x] = table[idx * Cc + threadIdx.x];
}

// x[b,l,c] += sum_m xpair[b,l,m,c] / valid[b]
__global__ void k_distsum(const float* __restrict__ xpair, const unsigned char* __restrict__ mask,
                          float* __restrict__ x) {
    const int bl = blockIdx.x, b = bl / Lc, c = threadIdx.x;
    float s = 0.0f;
    const float* base = xpair + (size_t)bl * Lc * Cc + c;
    for (int m = 0; m < Lc; ++m) s += base[(size_t)m * Cc];
    __shared__ int vc;
    if (c == 0) { int t = 0; for (int m = 0; m < Lc; ++m) t += mask[b * Lc + m] ? 0 : 1; vc = t; }
    __syncthreads();
    x[(size_t)bl * Cc + c] += s / (float)vc;
}

// ---------------- pair bias: pb[b,h,l,m] = sum_c xpair[b,l,m,c]*Wpb[c,h] ----------------
__global__ void k_pairbias(const float* __restrict__ xpair, const float* __restrict__ Wpb,
                           float* __restrict__ pb) {
    const int wid = blockIdx.x * 8 + (threadIdx.x >> 5);
    const int lane = threadIdx.x & 31;
    const float* row = xpair + (size_t)wid * Cc;
    float part[Hc];
#pragma unroll
    for (int h = 0; h < Hc; ++h) part[h] = 0.0f;
    for (int c = lane; c < Cc; c += 32) {
        float xv = row[c];
#pragma unroll
        for (int h = 0; h < Hc; ++h) part[h] += xv * Wpb[c * Hc + h];
    }
#pragma unroll
    for (int h = 0; h < Hc; ++h) {
        float v = part[h];
        for (int off = 16; off > 0; off >>= 1) v += __shfl_down(v, off, 32);
        if (lane == 0) {
            int m = wid % Lc, l = (wid / Lc) % Lc, b = wid / (Lc * Lc);
            pb[(((size_t)b * Hc + h) * Lc + l) * Lc + m] = v;
        }
    }
}

// ---------------- fused attention (softmax over L=256) ----------------
__global__ void k_attn(const float* __restrict__ q, const float* __restrict__ k,
                       const float* __restrict__ v, const float* __restrict__ bias,
                       const float* __restrict__ pb, const unsigned char* __restrict__ mask,
                       float* __restrict__ o) {
    const int id = blockIdx.x;                 // (b,h,l)
    const int l = id % Lc, h = (id / Lc) % Hc, b = id / (Lc * Hc);
    const int m = threadIdx.x;
    __shared__ float sp[Lc];
    __shared__ float red[Lc];
    const float* qrow = q + ((size_t)(b * Lc + l) * Cc + h * DHc);
    const float* krow = k + ((size_t)(b * Lc + m) * Cc + h * DHc);
    float s = 0.0f;
#pragma unroll
    for (int d = 0; d < DHc; ++d) s += qrow[d] * krow[d];
    s = s * 0.25f + bias[(size_t)(b * Lc + l) * Lc + m]
                  + pb[(((size_t)b * Hc + h) * Lc + l) * Lc + m];
    if (mask[b * Lc + m]) s = -1e9f;
    red[m] = s; __syncthreads();
    for (int off = 128; off > 0; off >>= 1) { if (m < off) red[m] = fmaxf(red[m], red[m + off]); __syncthreads(); }
    float mx = red[0]; __syncthreads();
    float e = __expf(s - mx);
    red[m] = e; sp[m] = e; __syncthreads();
    for (int off = 128; off > 0; off >>= 1) { if (m < off) red[m] += red[m + off]; __syncthreads(); }
    float inv = 1.0f / red[0]; __syncthreads();
    const int d = threadIdx.x & 15, g = threadIdx.x >> 4;
    float acc = 0.0f;
    for (int mm = g; mm < Lc; mm += 16) acc += sp[mm] * v[(size_t)(b * Lc + mm) * Cc + h * DHc + d];
    red[threadIdx.x] = acc; __syncthreads();
    for (int off = 128; off >= 16; off >>= 1) { if (threadIdx.x < off) red[threadIdx.x] += red[threadIdx.x + off]; __syncthreads(); }
    if (threadIdx.x < 16) o[(size_t)(b * Lc + l) * Cc + h * DHc + threadIdx.x] = red[threadIdx.x] * inv;
}

// ---------------- final masking / pooling / heads ----------------
__global__ void k_pairmaskb(const float* __restrict__ xpair, const unsigned char* __restrict__ mask,
                            __bf16* __restrict__ dst) {
    for (size_t i = (size_t)blockIdx.x * blockDim.x + threadIdx.x; i < PAIRc;
         i += (size_t)gridDim.x * blockDim.x) {
        const size_t lm = i >> 7;
        const int m = (int)(lm % Lc);
        const size_t bl = lm / Lc;
        const int l = (int)(bl % Lc), b = (int)(bl / Lc);
        float v = (mask[b * Lc + l] || mask[b * Lc + m]) ? 0.0f : xpair[i];
        dst[i] = f2bf(v);
    }
}
__global__ void k_pool(const float* __restrict__ xpair, const unsigned char* __restrict__ mask,
                       float* __restrict__ pool) {
    const int bl = blockIdx.x, b = bl / Lc, l = bl % Lc, c = threadIdx.x;
    const bool padl = mask[b * Lc + l] != 0;
    float s = 0.0f;
    for (int m = 0; m < Lc; ++m)
        if (!(padl || mask[b * Lc + m])) s += xpair[((size_t)bl * Lc + m) * Cc + c];
    __shared__ int vc;
    if (c == 0) { int t = 0; for (int m = 0; m < Lc; ++m) t += mask[b * Lc + m] ? 0 : 1; vc = t; }
    __syncthreads();
    pool[(size_t)bl * Cc + c] = s / (float)vc;
}
__global__ void k_plddt(const __bf16* __restrict__ t, const float* __restrict__ W2,
                        const float* __restrict__ b2, float* __restrict__ out) {
    const int bl = blockIdx.x, nb = threadIdx.x;
    if (nb >= NBc) return;
    float s = b2[nb];
    for (int c = 0; c < Cc; ++c) s += bf2f(t[(size_t)bl * Cc + c]) * W2[c * NBc + nb];
    out[(size_t)bl * NBc + nb] = s;
}

// ================= host-side schedule =================
extern "C" void kernel_launch(void* const* d_in, const int* in_sizes, int n_in,
                              void* d_out, int out_size, void* d_ws, size_t ws_size,
                              hipStream_t stream) {
    (void)in_sizes; (void)n_in; (void)out_size; (void)ws_size;
    const float* dx   = (const float*)d_in[0];
    const float* pos  = (const float*)d_in[1];
    const unsigned char* mask = (const unsigned char*)d_in[2];
    const float* W_s  = (const float*)d_in[3];
    const float* b_s  = (const float*)d_in[4];
    const float* dtab = (const float*)d_in[5];
    const float* W_ps1= (const float*)d_in[6];
    const float* b_ps1= (const float*)d_in[7];
    const float* W_ps2= (const float*)d_in[8];
    const float* b_ps2= (const float*)d_in[9];
    const float* W_z  = (const float*)d_in[10];
    const float* b_z  = (const float*)d_in[11];
    const float* ln1_g=(const float*)d_in[12]; const float* ln1_b=(const float*)d_in[13];
    const float* Wq=(const float*)d_in[14];    const float* bq=(const float*)d_in[15];
    const float* Wk=(const float*)d_in[16];    const float* bk=(const float*)d_in[17];
    const float* Wv=(const float*)d_in[18];    const float* bv=(const float*)d_in[19];
    const float* Wo=(const float*)d_in[20];    const float* bo=(const float*)d_in[21];
    const float* Wpb=(const float*)d_in[22];
    const float* ln2_g=(const float*)d_in[23]; const float* ln2_b=(const float*)d_in[24];
    const float* Wf1=(const float*)d_in[25];   const float* bf1=(const float*)d_in[26];
    const float* Wf2=(const float*)d_in[27];   const float* bf2=(const float*)d_in[28];
    const float* Wa=(const float*)d_in[29];    const float* ba=(const float*)d_in[30];
    const float* Wb=(const float*)d_in[31];    const float* bb=(const float*)d_in[32];
    const float* lnp_g=(const float*)d_in[33]; const float* lnp_b=(const float*)d_in[34];
    const float* Wp1=(const float*)d_in[35];   const float* bp1=(const float*)d_in[36];
    const float* Wp2=(const float*)d_in[37];   const float* bp2=(const float*)d_in[38];

    // -------- workspace carve-out --------
    char* base = (char*)d_ws; size_t off = 0;
    auto alloc = [&](size_t bytes) -> void* {
        void* p = base + off; off = (off + bytes + 255) & ~(size_t)255; return p;
    };
    float*  xpair = (float*) alloc(PAIRc * 4);
    __bf16* pairA = (__bf16*)alloc(PAIRc * 2);
    __bf16* pairB = (__bf16*)alloc(PAIRc * 2);
    float*  bias  = (float*) alloc((size_t)BLLc * 4);
    float*  pb    = (float*) alloc((size_t)Bc * Hc * Lc * Lc * 4);
    const size_t BLC = (size_t)BLc * Cc;
    float*  x    = (float*) alloc(BLC * 4);
    float*  q    = (float*) alloc(BLC * 4);
    float*  k    = (float*) alloc(BLC * 4);
    float*  v    = (float*) alloc(BLC * 4);
    float*  o    = (float*) alloc(BLC * 4);
    float*  a    = (float*) alloc(BLC * 4);
    float*  bpp  = (float*) alloc(BLC * 4);
    float*  pool = (float*) alloc(BLC * 4);
    __bf16* hbf  = (__bf16*)alloc(BLC * 2);
    __bf16* actbf= (__bf16*)alloc(BLC * 2);
    __bf16* tbf  = (__bf16*)alloc(BLC * 2);
    __bf16* xinbf= (__bf16*)alloc((size_t)BLc * CINc * 2);
    __bf16* wst  = (__bf16*)alloc((size_t)CINc * Cc * 2);
    const size_t WCC = (size_t)NLAYERc * Cc * Cc;
    __bf16* wqt  = (__bf16*)alloc(WCC * 2);
    __bf16* wkt  = (__bf16*)alloc(WCC * 2);
    __bf16* wvt  = (__bf16*)alloc(WCC * 2);
    __bf16* wot  = (__bf16*)alloc(WCC * 2);
    __bf16* wf1t = (__bf16*)alloc(WCC * 2);
    __bf16* wf2t = (__bf16*)alloc(WCC * 2);
    __bf16* wat  = (__bf16*)alloc(WCC * 2);
    __bf16* wbt  = (__bf16*)alloc(WCC * 2);
    __bf16* wp1t = (__bf16*)alloc(WCC * 2);
    __bf16* wp2t = (__bf16*)alloc(WCC * 2);
    __bf16* wps1t= (__bf16*)alloc((size_t)Cc * Cc * 2);
    __bf16* wzt  = (__bf16*)alloc((size_t)Cc * NZc * 2);

    float* out_plddt = (float*)d_out;
    float* out_pde   = (float*)d_out + (size_t)Bc * Lc * NBc;

    // -------- weight conversion (f32 -> bf16, transposed [N,K]) --------
    k_wt<<<2048, 256, 0, stream>>>(W_s,  wst,  CINc, Cc, 1);
    k_wt<<<2048, 256, 0, stream>>>(W_ps1,wps1t,Cc,   Cc, 1);
    k_wt<<<2048, 256, 0, stream>>>(W_z,  wzt,  Cc,  NZc, 1);
    k_wt<<<2048, 256, 0, stream>>>(Wq,  wqt,  Cc, Cc, NLAYERc);
    k_wt<<<2048, 256, 0, stream>>>(Wk,  wkt,  Cc, Cc, NLAYERc);
    k_wt<<<2048, 256, 0, stream>>>(Wv,  wvt,  Cc, Cc, NLAYERc);
    k_wt<<<2048, 256, 0, stream>>>(Wo,  wot,  Cc, Cc, NLAYERc);
    k_wt<<<2048, 256, 0, stream>>>(Wf1, wf1t, Cc, Cc, NLAYERc);
    k_wt<<<2048, 256, 0, stream>>>(Wf2, wf2t, Cc, Cc, NLAYERc);
    k_wt<<<2048, 256, 0, stream>>>(Wa,  wat,  Cc, Cc, NLAYERc);
    k_wt<<<2048, 256, 0, stream>>>(Wb,  wbt,  Cc, Cc, NLAYERc);
    k_wt<<<2048, 256, 0, stream>>>(Wp1, wp1t, Cc, Cc, NLAYERc);
    k_wt<<<2048, 256, 0, stream>>>(Wp2, wp2t, Cc, Cc, NLAYERc);

    // -------- input projection: x = dx @ W_s + b_s --------
    k_f2b<<<2048, 256, 0, stream>>>(dx, xinbf, (size_t)BLc * CINc);
    k_gemm<8, 0><<<BLc / 128, 256, 0, stream>>>(xinbf, wst, b_s, nullptr, x, nullptr, CINc);

    // -------- distance embedding + pair init --------
    k_dist<<<BLLc, Cc, 0, stream>>>(pos, dtab, xpair, bias);
    k_distsum<<<BLc, Cc, 0, stream>>>(xpair, mask, x);

    // -------- layers --------
    for (int i = 0; i < NLAYERc; ++i) {
        const size_t wcc = (size_t)i * Cc * Cc;
        const size_t wc  = (size_t)i * Cc;
        // attention block
        k_ln_bf16<<<BLc, Cc, 0, stream>>>(x, ln1_g + wc, ln1_b + wc, hbf);
        k_gemm<8, 0><<<BLc / 128, 256, 0, stream>>>(hbf, wqt + wcc, bq + wc, nullptr, q, nullptr, Cc);
        k_gemm<8, 0><<<BLc / 128, 256, 0, stream>>>(hbf, wkt + wcc, bk + wc, nullptr, k, nullptr, Cc);
        k_gemm<8, 0><<<BLc / 128, 256, 0, stream>>>(hbf, wvt + wcc, bv + wc, nullptr, v, nullptr, Cc);
        k_pairbias<<<BLLc / 8, 256, 0, stream>>>(xpair, Wpb + (size_t)i * Cc * Hc, pb);
        k_attn<<<Bc * Hc * Lc, 256, 0, stream>>>(q, k, v, bias, pb, mask, o);
        k_f2b<<<512, 256, 0, stream>>>(o, actbf, BLC);
        k_gemm<8, 1><<<BLc / 128, 256, 0, stream>>>(actbf, wot + wcc, bo + wc, x, x, nullptr, Cc);
        // FFN
        k_ln_bf16<<<BLc, Cc, 0, stream>>>(x, ln2_g + wc, ln2_b + wc, hbf);
        k_gemm<8, 2><<<BLc / 128, 256, 0, stream>>>(hbf, wf1t + wcc, bf1 + wc, nullptr, nullptr, tbf, Cc);
        k_gemm<8, 1><<<BLc / 128, 256, 0, stream>>>(tbf, wf2t + wcc, bf2 + wc, x, x, nullptr, Cc);
        // pair update: a, bpp from post-FFN x
        k_f2b<<<512, 256, 0, stream>>>(x, actbf, BLC);
        k_gemm<8, 0><<<BLc / 128, 256, 0, stream>>>(actbf, wat + wcc, ba + wc, nullptr, a,   nullptr, Cc);
        k_gemm<8, 0><<<BLc / 128, 256, 0, stream>>>(actbf, wbt + wcc, bb + wc, nullptr, bpp, nullptr, Cc);
        // fused rank-1 update + pair LN -> bf16
        k_pairupd_ln<<<BLLc / 8, 256, 0, stream>>>(xpair, a, bpp, lnp_g + wc, lnp_b + wc, pairA);
        // pair FFN (TDM-staged weights)
        k_gemm_tdm<8, 2><<<BLLc / 128, 256, 0, stream>>>(pairA, wp1t + wcc, bp1 + wc, nullptr, nullptr, pairB);
        k_gemm_tdm<8, 1><<<BLLc / 128, 256, 0, stream>>>(pairB, wp2t + wcc, bp2 + wc, xpair, xpair, nullptr);
    }

    // -------- heads --------
    k_pairmaskb<<<4096, 256, 0, stream>>>(xpair, mask, pairA);
    k_gemm_tdm<4, 0><<<BLLc / 128, 256, 0, stream>>>(pairA, wzt, b_z, nullptr, out_pde, nullptr);
    k_pool<<<BLc, Cc, 0, stream>>>(xpair, mask, pool);
    k_f2b<<<256, 256, 0, stream>>>(pool, actbf, BLC);
    k_gemm<8, 2><<<BLc / 128, 256, 0, stream>>>(actbf, wps1t, b_ps1, nullptr, nullptr, tbf, Cc);
    k_plddt<<<BLc, 64, 0, stream>>>(tbf, W_ps2, b_ps2, out_plddt);
}
